// DGConv_77429670412662
// MI455X (gfx1250) — matrix operand order repaired and checked by reading, not compiled, not run on hardware
//
#include <hip/hip_runtime.h>
#include <hip/hip_bf16.h>

// ---------------------------------------------------------------------------
// DGConv (EdgeConv-style) fused pipeline for MI455X (gfx1250, wave32, WMMA).
//   B=4, C=64, N=4096, K=20, OUT_C=64
// Kernels:
//   1) prep:    x (B,C,N) -> xt_f16 (B,N,C), sq[b,n] (fp32)
//   2) wconv:   W -> W1_f16 (o,c), Wd_f16 = (W2-W1) (o,c)
//   3) knn:     fused WMMA distance GEMM (16 rows x 4096 cols in LDS) +
//               register-buffered top-20 (single LDS scan + shuffle argmax)
//   4) gp:      WMMA  g = xt*W1^T,  p2 = xt*(W2-W1)^T
//   5) gather:  out_pre[b,o,n] = max_k g[idx_k,o] + p2[n,o]; atomic BN stats
//   6) bn:      normalize with gamma/beta
// ---------------------------------------------------------------------------

#define BATCH 4
#define CHN   64
#define NPTS  4096
#define KNN   20
#define OUTC  64
#define NPOINTS (BATCH * NPTS)   // 16384

typedef _Float16 v16h __attribute__((ext_vector_type(16)));
typedef _Float16 v8h  __attribute__((ext_vector_type(8)));
typedef float    v8f  __attribute__((ext_vector_type(8)));

#define WMMA_F16F32(A, B, C) \
  __builtin_amdgcn_wmma_f32_16x16x32_f16(false, (A), false, (B), (short)0, (C), false, false)

// Sorted insert into per-lane top-4 (val desc, idx asc on ties via strict >).
#define TOP4_INSERT(v, j)                                                     \
  if ((v) > v3) {                                                             \
    if ((v) > v0)      { v3=v2;i3=i2; v2=v1;i2=i1; v1=v0;i1=i0; v0=(v);i0=(j); } \
    else if ((v) > v1) { v3=v2;i3=i2; v2=v1;i2=i1; v1=(v);i1=(j); }           \
    else if ((v) > v2) { v3=v2;i3=i2; v2=(v);i2=(j); }                        \
    else               { v3=(v); i3=(j); }                                    \
  }

// --- CDNA5 16-bit A fragment (16x32, MxK), row-major source, row stride = 64
// lane l (l<16): row M=l,  K = k0+0..15
// lane l (>=16): row M=l-16, K = k0+16..31
__device__ __forceinline__ v16h load_a_frag(const _Float16* base, int k0, int lane) {
  int r  = lane & 15;
  int hi = lane >> 4;
  const _Float16* p = base + r * 64 + k0 + hi * 16;
  return *(const v16h*)(p);
}

// --- CDNA5 16-bit B fragment (32x16, KxN). Source stored as one contiguous
// 64-element K-vector per column (stride 64): lanes 0-15 hold K=k0..k0+15 of
// column l, lanes 16-31 hold K=k0+16..k0+31 of column l-16.
__device__ __forceinline__ v16h load_b_frag(const _Float16* colbase, int k0, int lane) {
  int c  = lane & 15;
  int hi = lane >> 4;
  return *(const v16h*)(colbase + c * 64 + k0 + hi * 16);
}

// ---------------------------------------------------------------------------
__global__ void dg_prep(const float* __restrict__ x, _Float16* __restrict__ xtf,
                        float* __restrict__ sq) {
  int P = blockIdx.x * blockDim.x + threadIdx.x;   // 0..16383
  int b = P >> 12, n = P & (NPTS - 1);
  const float* xp = x + (size_t)b * CHN * NPTS + n;
  _Float16* op = xtf + (size_t)P * CHN;
  float s = 0.f;
#pragma unroll
  for (int c = 0; c < CHN; ++c) {
    float v = xp[(size_t)c * NPTS];
    s += v * v;
    op[c] = (_Float16)v;
  }
  sq[P] = s;
}

__global__ void dg_wconv(const float* __restrict__ W, _Float16* __restrict__ w1f,
                         _Float16* __restrict__ wdf) {
  int i = blockIdx.x * blockDim.x + threadIdx.x;   // 0..4095
  int o = i >> 6, c = i & 63;
  float a = W[o * 128 + c];
  float b = W[o * 128 + 64 + c];
  w1f[i] = (_Float16)a;
  wdf[i] = (_Float16)(b - a);
}

// ---------------------------------------------------------------------------
// Fused distance-GEMM + top-K. One block = 16 rows; dynamic LDS 16x4096 f32.
// Ranking key = inner(n,m) - 0.5*sq[m]  (row term -sq[n] is rank-invariant).
__global__ __launch_bounds__(512) void dg_knn(const _Float16* __restrict__ xtf,
                                              const float* __restrict__ sq,
                                              int* __restrict__ idxout) {
  extern __shared__ float tile[];                  // 16 * 4096 floats
  int b  = blockIdx.x >> 8;                        // 256 row-tiles per batch
  int r0 = (blockIdx.x & 255) << 4;
  const _Float16* xb = xtf + (size_t)b * NPTS * CHN;
  int tid = threadIdx.x, lane = tid & 31, wave = tid >> 5;  // 16 waves
  int cN = lane & 15, hi = lane >> 4;

  // A tile (16 rows x 64) kept in registers for all 256 column tiles.
  v16h a0 = load_a_frag(xb + (size_t)r0 * CHN, 0, lane);
  v16h a1 = load_a_frag(xb + (size_t)r0 * CHN, 32, lane);

  for (int t = wave; t < NPTS / 16; t += 16) {
    int m0 = t << 4;
    v16h b0 = load_b_frag(xb + (size_t)m0 * CHN, 0, lane);
    v16h b1 = load_b_frag(xb + (size_t)m0 * CHN, 32, lane);
    v8f acc = {};
    acc = WMMA_F16F32(a0, b0, acc);
    acc = WMMA_F16F32(a1, b1, acc);
    float sqc = sq[b * NPTS + m0 + cN];
#pragma unroll
    for (int i = 0; i < 8; ++i) {                  // D: M = i + 8*hi, N = cN
      tile[(i + hi * 8) * NPTS + m0 + cN] = acc[i] - 0.5f * sqc;
    }
  }
  __syncthreads();

  // ---- Top-20 per row: wave w owns row w. ----
  // Single conflict-free scan; each lane keeps a sorted top-4 of its stripe
  // (j == lane mod 32) in registers. 20 rounds of shuffle-argmax over heads.
  const float NEG = -3.0e38f;
  float* row = tile + wave * NPTS;
  int gr = b * NPTS + r0 + wave;

  float v0 = NEG, v1 = NEG, v2 = NEG, v3 = NEG;
  int   i0 = 0,   i1 = 0,   i2 = 0,   i3 = 0;
  for (int j = lane; j < NPTS; j += 32) {
    float v = row[j];
    TOP4_INSERT(v, j)
  }

  for (int kk = 0; kk < KNN; ++kk) {
    float bv = v0;
    int   bi = i0;
#pragma unroll
    for (int off = 16; off; off >>= 1) {
      float ov = __shfl_xor(bv, off, 32);
      int   oi = __shfl_xor(bi, off, 32);
      if (ov > bv || (ov == bv && oi < bi)) { bv = ov; bi = oi; }
    }
    if (lane == 0) {
      idxout[(size_t)gr * KNN + kk] = bi;
      row[bi] = NEG;                       // mark consumed (for rare refill)
    }
    if ((bi & 31) == lane) {               // owner lane pops its buffer
      v0 = v1; i0 = i1; v1 = v2; i1 = i2; v2 = v3; i2 = i3; v3 = NEG; i3 = 0;
      if (v0 == NEG) {                     // buffer exhausted (rare): rescan
        for (int j = lane; j < NPTS; j += 32) {
          float v = row[j];                // consumed slots read back as NEG
          TOP4_INSERT(v, j)
        }
      }
    }
  }
}

// ---------------------------------------------------------------------------
// g = xt * W1^T, p2 = xt * (W2 - W1)^T   (both (B,N,64), fp32)
__global__ __launch_bounds__(256) void dg_gp(const _Float16* __restrict__ xtf,
                                             const _Float16* __restrict__ w1f,
                                             const _Float16* __restrict__ wdf,
                                             float* __restrict__ g,
                                             float* __restrict__ p2) {
  int lane = threadIdx.x & 31, wave = threadIdx.x >> 5;
  int tileId = blockIdx.x * 8 + wave;              // 1024 row-tiles total
  int b  = tileId >> 8;
  int r0 = (tileId & 255) << 4;
  const _Float16* xb = xtf + (size_t)b * NPTS * CHN;
  int cN = lane & 15, hi = lane >> 4;

  v16h a0 = load_a_frag(xb + (size_t)r0 * CHN, 0, lane);
  v16h a1 = load_a_frag(xb + (size_t)r0 * CHN, 32, lane);

#pragma unroll
  for (int o0 = 0; o0 < OUTC; o0 += 16) {
    v16h b0 = load_b_frag(w1f + o0 * 64, 0, lane);
    v16h b1 = load_b_frag(w1f + o0 * 64, 32, lane);
    v8f acc = {};
    acc = WMMA_F16F32(a0, b0, acc);
    acc = WMMA_F16F32(a1, b1, acc);
#pragma unroll
    for (int i = 0; i < 8; ++i) {
      int rr = r0 + i + hi * 8;
      g[((size_t)b * NPTS + rr) * OUTC + o0 + cN] = acc[i];
    }
    v16h c0 = load_b_frag(wdf + o0 * 64, 0, lane);
    v16h c1 = load_b_frag(wdf + o0 * 64, 32, lane);
    v8f ac2 = {};
    ac2 = WMMA_F16F32(a0, c0, ac2);
    ac2 = WMMA_F16F32(a1, c1, ac2);
#pragma unroll
    for (int i = 0; i < 8; ++i) {
      int rr = r0 + i + hi * 8;
      p2[((size_t)b * NPTS + rr) * OUTC + o0 + cN] = ac2[i];
    }
  }
}

// ---------------------------------------------------------------------------
// out_pre[b,o,n] = max_k g[idx_k, o] + p2[n, o]; accumulate BN sums.
__global__ __launch_bounds__(256) void dg_gather(const float* __restrict__ g,
                                                 const float* __restrict__ p2,
                                                 const int* __restrict__ idx,
                                                 float* __restrict__ outpre,
                                                 float* __restrict__ sums) {
  __shared__ int   sidx[4][KNN];
  __shared__ float sv[256], sv2[256];
  int tid = threadIdx.x;
  int o = tid & 63, pi = tid >> 6;                 // 64 channels x 4 points
  int P = blockIdx.x * 4 + pi;
  int b = P >> 12, n = P & (NPTS - 1);
  if (o < KNN) sidx[pi][o] = idx[(size_t)P * KNN + o];
  __syncthreads();

  const float* gb = g + (size_t)b * NPTS * OUTC;
  // Warm the irregular g rows into cache (global_prefetch_b8 path).
  __builtin_prefetch(&gb[(size_t)sidx[pi][o % KNN] * OUTC + o], 0, 3);
  float m = -3.0e38f;
#pragma unroll
  for (int k = 0; k < KNN; ++k) {
    int j = sidx[pi][k];
    m = fmaxf(m, gb[(size_t)j * OUTC + o]);
  }
  float res = m + p2[(size_t)P * OUTC + o];
  outpre[((size_t)b * OUTC + o) * NPTS + n] = res;

  sv[tid] = res;
  sv2[tid] = res * res;
  __syncthreads();
  if (tid < 64) {
    float s  = sv[tid]  + sv[tid + 64]  + sv[tid + 128]  + sv[tid + 192];
    float s2 = sv2[tid] + sv2[tid + 64] + sv2[tid + 128] + sv2[tid + 192];
    atomicAdd(&sums[o], s);
    atomicAdd(&sums[OUTC + o], s2);
  }
}

__global__ void dg_bn(const float* __restrict__ outpre, const float* __restrict__ sums,
                      const float* __restrict__ gamma, const float* __restrict__ beta,
                      float* __restrict__ out) {
  int i = blockIdx.x * blockDim.x + threadIdx.x;   // B*OUTC*NPTS elements
  int o = (i >> 12) & 63;
  const float invN = 1.0f / (float)(BATCH * NPTS);
  float mean = sums[o] * invN;
  float var  = sums[OUTC + o] * invN - mean * mean;
  float inv  = rsqrtf(var + 1e-5f);
  out[i] = (outpre[i] - mean) * inv * gamma[o] + beta[o];
}

// ---------------------------------------------------------------------------
// Workspace layout (bytes), ~16.1 MB total
static constexpr size_t OFF_XTF = 0;                           // 2,097,152 (f16)
static constexpr size_t OFF_SQ  = OFF_XTF + (size_t)NPOINTS * CHN * 2;   // 65,536
static constexpr size_t OFF_W1  = OFF_SQ  + (size_t)NPOINTS * 4;         // 8,192
static constexpr size_t OFF_WD  = OFF_W1  + (size_t)OUTC * CHN * 2;      // 8,192
static constexpr size_t OFF_G   = OFF_WD  + (size_t)OUTC * CHN * 2;      // 4,194,304
static constexpr size_t OFF_P2  = OFF_G   + (size_t)NPOINTS * OUTC * 4;  // 4,194,304
static constexpr size_t OFF_IDX = OFF_P2  + (size_t)NPOINTS * OUTC * 4;  // 1,310,720
static constexpr size_t OFF_OP  = OFF_IDX + (size_t)NPOINTS * KNN * 4;   // 4,194,304
static constexpr size_t OFF_SUM = OFF_OP  + (size_t)NPOINTS * OUTC * 4;  // 512

extern "C" void kernel_launch(void* const* d_in, const int* in_sizes, int n_in,
                              void* d_out, int out_size, void* d_ws, size_t ws_size,
                              hipStream_t stream) {
  (void)in_sizes; (void)n_in; (void)out_size; (void)ws_size;
  const float* x     = (const float*)d_in[0];
  const float* W     = (const float*)d_in[1];
  const float* gamma = (const float*)d_in[2];
  const float* beta  = (const float*)d_in[3];
  float* out = (float*)d_out;

  char* ws = (char*)d_ws;
  _Float16* xtf  = (_Float16*)(ws + OFF_XTF);
  float*    sq   = (float*)(ws + OFF_SQ);
  _Float16* w1f  = (_Float16*)(ws + OFF_W1);
  _Float16* wdf  = (_Float16*)(ws + OFF_WD);
  float*    g    = (float*)(ws + OFF_G);
  float*    p2   = (float*)(ws + OFF_P2);
  int*      idxb = (int*)(ws + OFF_IDX);
  float*    opre = (float*)(ws + OFF_OP);
  float*    sums = (float*)(ws + OFF_SUM);

  dg_prep<<<NPOINTS / 256, 256, 0, stream>>>(x, xtf, sq);
  dg_wconv<<<(OUTC * CHN) / 256, 256, 0, stream>>>(W, w1f, wdf);
  // 16 rows x 4096 cols f32 tile in dynamic LDS = 256 KB (WGP has 320 KB)
  dg_knn<<<BATCH * (NPTS / 16), 512, 16 * NPTS * sizeof(float), stream>>>(xtf, sq, idxb);
  dg_gp<<<NPOINTS / 128, 256, 0, stream>>>(xtf, w1f, wdf, g, p2);
  hipMemsetAsync(sums, 0, 2 * OUTC * sizeof(float), stream);
  dg_gather<<<NPOINTS / 4, 256, 0, stream>>>(g, p2, idxb, opre, sums);
  dg_bn<<<(BATCH * OUTC * NPTS) / 256, 256, 0, stream>>>(opre, sums, gamma, beta, out);
}
